// Itloss_jiasaw_18433999635079
// MI455X (gfx1250) — matrix operand order repaired and checked by reading, not compiled, not run on hardware
//
#include <hip/hip_runtime.h>
#include <hip/hip_bf16.h>
#include <math.h>
#include <stdint.h>

typedef __attribute__((ext_vector_type(16))) __bf16 v16bf;
typedef __attribute__((ext_vector_type(8)))  float  v8f;
typedef int v4i_t __attribute__((vector_size(4 * sizeof(int))));  // matches builtin param

#define BB     32768      // batch
#define DD     512        // feature dim
#define PP     999        // text prototypes
#define PPAD   1008       // 63 * 16 padded proto rows
#define NT     63         // 16-wide column tiles covering PPAD
#define NCLS1  1001       // logits width (NUM_CLASS + 1)
#define INVT   10.0f      // 1 / TEMPERATURE

// gfx1250 async global->LDS path (fallback to plain loads if builtin missing)
#if defined(__has_builtin)
# if __has_builtin(__builtin_amdgcn_global_load_async_to_lds_b128)
#  define HAVE_ASYNC_LDS 1
# endif
#endif
#ifndef HAVE_ASYNC_LDS
# define HAVE_ASYNC_LDS 0
#endif

#if defined(__has_builtin)
# if __has_builtin(__builtin_amdgcn_s_wait_asynccnt)
#  define WAIT_ASYNC() __builtin_amdgcn_s_wait_asynccnt(0)
# endif
#endif
#ifndef WAIT_ASYNC
# define WAIT_ASYNC() asm volatile("s_wait_asynccnt 0x0" ::: "memory")
#endif

// e^(10x) from e = e^x via squarings (saves a transcendental on the trans pipe,
// which co-issues with WMMA)
__device__ __forceinline__ float pow10f_(float e) {
  const float e2 = e * e;
  const float e4 = e2 * e2;
  const float e8 = e4 * e4;
  return e8 * e2;
}

// acc layout in d_ws: [0]=simSum [1]=sepSum [2]=jigCount [3]=ceSum [4]=textSum

// ---------------- kernel 0: zero accumulators ----------------
__global__ void init_kernel(float* __restrict__ acc) {
  if (threadIdx.x < 8) acc[threadIdx.x] = 0.0f;
}

// ---- kernel 1: normalize prototypes, split f32 -> bf16 hi/lo, pad to 1008 rows ----
__global__ __launch_bounds__(256) void norm_kernel(const float* __restrict__ tp,
                                                   __bf16* __restrict__ pH,
                                                   __bf16* __restrict__ pL) {
  __shared__ float red[8];
  const int p = blockIdx.x;
  const int tid = threadIdx.x;
  const size_t base = (size_t)p * DD;
  if (p < PP) {
    const float x0 = tp[base + tid];
    const float x1 = tp[base + tid + 256];
    float s = x0 * x0 + x1 * x1;
#pragma unroll
    for (int m = 1; m < 32; m <<= 1) s += __shfl_xor(s, m, 32);
    if ((tid & 31) == 0) red[tid >> 5] = s;
    __syncthreads();
    float S = 0.f;
#pragma unroll
    for (int i = 0; i < 8; ++i) S += red[i];
    const float rn = 1.0f / sqrtf(S);
    const float v0 = x0 * rn, v1 = x1 * rn;
    const __bf16 h0 = (__bf16)v0, h1 = (__bf16)v1;
    pH[base + tid]       = h0;
    pH[base + tid + 256] = h1;
    pL[base + tid]       = (__bf16)(v0 - (float)h0);
    pL[base + tid + 256] = (__bf16)(v1 - (float)h1);
  } else {  // zero padding rows (masked out later, but loads must be benign)
    pH[base + tid]       = (__bf16)0.f;
    pH[base + tid + 256] = (__bf16)0.f;
    pL[base + tid]       = (__bf16)0.f;
    pL[base + tid + 256] = (__bf16)0.f;
  }
}

// ---------------- kernel 2: cross entropy over logits [B, 1001] ----------------
__global__ __launch_bounds__(256) void ce_kernel(const float* __restrict__ logits,
                                                 const int* __restrict__ labels,
                                                 float* __restrict__ acc) {
  __shared__ float buf[NCLS1];
  __shared__ float red[8];
  const int b = blockIdx.x;
  const int tid = threadIdx.x;
  const float* row = logits + (size_t)b * NCLS1;
  for (int i = tid; i < NCLS1; i += 256) buf[i] = row[i];
  __syncthreads();
  float m = -INFINITY;
  for (int i = tid; i < NCLS1; i += 256) m = fmaxf(m, buf[i]);
#pragma unroll
  for (int q = 1; q < 32; q <<= 1) m = fmaxf(m, __shfl_xor(m, q, 32));
  if ((tid & 31) == 0) red[tid >> 5] = m;
  __syncthreads();
#pragma unroll
  for (int i = 0; i < 8; ++i) m = fmaxf(m, red[i]);   // block max in every thread
  __syncthreads();                                    // before red reuse
  float s = 0.f;
  for (int i = tid; i < NCLS1; i += 256) s += expf(buf[i] - m);
#pragma unroll
  for (int q = 1; q < 32; q <<= 1) s += __shfl_xor(s, q, 32);
  if ((tid & 31) == 0) red[tid >> 5] = s;
  __syncthreads();
  if (tid == 0) {
    float S = 0.f;
#pragma unroll
    for (int i = 0; i < 8; ++i) S += red[i];
    atomicAdd(&acc[3], m + logf(S) - buf[labels[b]]);
  }
}

// ------- kernel 3: fused features@protosT GEMM + row softmax stats (sim/sep) -------
__global__ __launch_bounds__(256) void sim_kernel(const float* __restrict__ feat,
                                                  const int* __restrict__ labels,
                                                  const __bf16* __restrict__ pH,
                                                  const __bf16* __restrict__ pL,
                                                  float* __restrict__ acc) {
  constexpr int LDA = 520;              // bf16 elements; 1040B row stride avoids bank conflicts
  __shared__ __bf16 aHi[16 * LDA];
  __shared__ __bf16 aLo[16 * LDA];
#if HAVE_ASYNC_LDS
  __shared__ float stage[8 * DD];       // f32 staging for async global->LDS copy
#endif
  __shared__ float wMax[8][16];
  __shared__ float wS1[8][16];
  __shared__ float wS2[8][16];
  __shared__ float wCap[8][16];         // per-wave captured raw[label] (0 if not seen)
  __shared__ int   lab[16];

  const int tid = threadIdx.x;
  const int rowBase = blockIdx.x * 16;

  if (tid < 16) { lab[tid] = labels[rowBase + tid]; }

#if HAVE_ASYNC_LDS
  // two 8-row phases: async-DMA f32 rows into LDS, then convert to bf16 hi/lo
  {
    const float* gsrc = feat + (size_t)rowBase * DD;
    for (int ph = 0; ph < 2; ++ph) {
      for (int e = tid; e < (8 * DD) / 4; e += 256) {
        const float* gp = gsrc + (size_t)ph * 8 * DD + e * 4;
        float* lp = &stage[e * 4];
        __builtin_amdgcn_global_load_async_to_lds_b128(
            (__attribute__((address_space(1))) v4i_t*)(uintptr_t)gp,
            (__attribute__((address_space(3))) v4i_t*)(uint32_t)(uintptr_t)lp,
            0, 0);
      }
      WAIT_ASYNC();
      __syncthreads();
      for (int e = tid; e < 8 * DD; e += 256) {
        const int r = (e >> 9) + ph * 8;
        const int k = e & (DD - 1);
        const float v = stage[e];
        const __bf16 h = (__bf16)v;
        aHi[r * LDA + k] = h;
        aLo[r * LDA + k] = (__bf16)(v - (float)h);
      }
      __syncthreads();
    }
  }
#else
  {
    const float4* f4 = (const float4*)(feat + (size_t)rowBase * DD);
    for (int e = tid; e < 16 * (DD / 4); e += 256) {
      const int r  = e / (DD / 4);
      const int k4 = (e % (DD / 4)) * 4;
      const float4 v = f4[e];
      const int bofs = r * LDA + k4;
      float vv[4] = { v.x, v.y, v.z, v.w };
#pragma unroll
      for (int i = 0; i < 4; ++i) {
        const __bf16 h = (__bf16)vv[i];
        aHi[bofs + i] = h;
        aLo[bofs + i] = (__bf16)(vv[i] - (float)h);
      }
    }
  }
  __syncthreads();
#endif

  const int w    = tid >> 5;
  const int lane = tid & 31;
  const int mloc = lane & 15;
  const int uh   = lane >> 4;          // half-wave id
  const int aoff = uh ? 8 : 0;         // A K sub-offset per ISA layout
  const int boff = uh ? 8 : 0;         // B K sub-offset per ISA layout

  // per-lane state for this lane's 8 rows (registers, no LDS in the tile loop):
  // online softmax (max, S1=sum e^{x/T}, S2=sum e^x) + captured raw[label]
  float lm[8], ls1[8], ls2[8], capt[8];
  int   labR[8];
#pragma unroll
  for (int r = 0; r < 8; ++r) {
    lm[r] = -INFINITY; ls1[r] = 0.f; ls2[r] = 0.f; capt[r] = 0.f;
    labR[r] = lab[r + (uh ? 8 : 0)];
  }

  for (int t = w; t < NT; t += 8) {
    const int pcol = t * 16 + mloc;    // proto column (< 1008 always)
    // prefetch next tile's B rows into cache (global_prefetch_b8)
    if (t + 8 < NT) {
      __builtin_prefetch(pH + (size_t)((t + 8) * 16 + mloc) * DD, 0, 1);
      __builtin_prefetch(pL + (size_t)((t + 8) * 16 + mloc) * DD, 0, 1);
    }
    const __bf16* __restrict__ bHrow = pH + (size_t)pcol * DD + boff;
    const __bf16* __restrict__ bLrow = pL + (size_t)pcol * DD + boff;
    const __bf16* __restrict__ aHrow = aHi + mloc * LDA + aoff;
    const __bf16* __restrict__ aLrow = aLo + mloc * LDA + aoff;
    v8f accv = {};
#pragma unroll 2
    for (int k0 = 0; k0 < DD; k0 += 32) {
      v16bf ah, al, bh, bl;
#pragma unroll
      for (int i = 0; i < 8; ++i) {
        ah[i] = aHrow[k0 + i];  ah[i + 8] = aHrow[k0 + 16 + i];
        al[i] = aLrow[k0 + i];  al[i + 8] = aLrow[k0 + 16 + i];
      }
#pragma unroll
      for (int i = 0; i < 8; ++i) { bh[i] = bHrow[k0 + i]; bh[i + 8] = bHrow[k0 + 16 + i]; bl[i] = bLrow[k0 + i]; bl[i + 8] = bLrow[k0 + 16 + i]; }
      accv = __builtin_amdgcn_wmma_f32_16x16x32_bf16(false, ah, false, bh, (short)0, accv, false, false);
      accv = __builtin_amdgcn_wmma_f32_16x16x32_bf16(false, ah, false, bl, (short)0, accv, false, false);
      accv = __builtin_amdgcn_wmma_f32_16x16x32_bf16(false, al, false, bh, (short)0, accv, false, false);
    }
    const bool colValid = (pcol < PP);
#pragma unroll
    for (int r = 0; r < 8; ++r) {
      float v = accv[r];
      if (colValid && labR[r] == pcol) capt[r] = v;   // v_cndmask, no LDS
      if (!colValid) v = -INFINITY;
      const float nM = fmaxf(lm[r], v);
      const float eo = expf(lm[r] - nM);   // rescale of old sums (1.0 if max unchanged)
      const float en = expf(v - nM);
      ls2[r] = ls2[r] * eo         + en;            // sum exp(raw - M)
      ls1[r] = ls1[r] * pow10f_(eo) + pow10f_(en);  // sum exp((raw - M)/T)
      lm[r] = nM;
    }
  }

  // one-time 16-lane-group merge of per-lane (max, S1, S2, capt)
#pragma unroll
  for (int r = 0; r < 8; ++r) {
#pragma unroll
    for (int msk = 1; msk < 16; msk <<= 1) {
      const float om  = __shfl_xor(lm[r],  msk, 32);
      const float os1 = __shfl_xor(ls1[r], msk, 32);
      const float os2 = __shfl_xor(ls2[r], msk, 32);
      capt[r] += __shfl_xor(capt[r], msk, 32);  // exactly one matcher block-wide
      const float nM = fmaxf(lm[r], om);
      const float so = expf(lm[r] - nM);
      const float sn = expf(om - nM);
      ls1[r] = ls1[r] * pow10f_(so) + os1 * pow10f_(sn);
      ls2[r] = ls2[r] * so          + os2 * sn;
      lm[r] = nM;
    }
    if (mloc == 0) {  // lane 0 -> rows 0..7, lane 16 -> rows 8..15 (disjoint)
      const int row = r + (uh ? 8 : 0);
      wMax[w][row] = lm[r];
      wS1[w][row]  = ls1[r];
      wS2[w][row]  = ls2[r];
      wCap[w][row] = capt[r];
    }
  }
  __syncthreads();

  if (tid < 32) {  // cross-wave merge + loss terms
    float sim = 0.f, sep = 0.f, cnt = 0.f;
    if (lane < 16) {
      const int row = lane;
      float M = -INFINITY;
#pragma unroll
      for (int ww = 0; ww < 8; ++ww) M = fmaxf(M, wMax[ww][row]);
      float S1 = 0.f, S2 = 0.f, rawv = 0.f;
#pragma unroll
      for (int ww = 0; ww < 8; ++ww) {
        S1 += wS1[ww][row] * expf((wMax[ww][row] - M) * INVT);
        S2 += wS2[ww][row] * expf(wMax[ww][row] - M);
        rawv += wCap[ww][row];
      }
      const int l = lab[row];
      if (l < PP)   sim = logf(S1) - (rawv - M) * INVT;  // -log_prob[label]
      if (l == 1000) { sep = M + logf(S2); cnt = 1.f; }  // logsumexp(raw)
    }
#pragma unroll
    for (int m = 1; m < 32; m <<= 1) {
      sim += __shfl_xor(sim, m, 32);
      sep += __shfl_xor(sep, m, 32);
      cnt += __shfl_xor(cnt, m, 32);
    }
    if (lane == 0) { atomicAdd(&acc[0], sim); atomicAdd(&acc[1], sep); atomicAdd(&acc[2], cnt); }
  }
}

// ------- kernel 4: textloss numerator  sum_{i!=j} exp(<p_i,p_j>)  via WMMA -------
__global__ __launch_bounds__(256) void text_kernel(const __bf16* __restrict__ pH,
                                                   const __bf16* __restrict__ pL,
                                                   float* __restrict__ acc) {
  __shared__ float red[8];
  const int tid = threadIdx.x;
  const int w = tid >> 5, lane = tid & 31;
  const int gw = blockIdx.x * 8 + w;
  float local = 0.f;
  if (gw < NT * NT) {
    const int ti = gw / NT, tj = gw % NT;
    const int mloc = lane & 15;
    const int uh = lane >> 4;
    const int aoff = uh ? 8 : 0;
    const int boff = uh ? 8 : 0;
    const __bf16* aH = pH + (size_t)(ti * 16 + mloc) * DD + aoff;
    const __bf16* aL = pL + (size_t)(ti * 16 + mloc) * DD + aoff;
    const __bf16* bH = pH + (size_t)(tj * 16 + mloc) * DD + boff;
    const __bf16* bL = pL + (size_t)(tj * 16 + mloc) * DD + boff;
    v8f accv = {};
    for (int k0 = 0; k0 < DD; k0 += 32) {
      v16bf ah, al, bh, bl;
#pragma unroll
      for (int i = 0; i < 8; ++i) {
        ah[i] = aH[k0 + i];  ah[i + 8] = aH[k0 + 16 + i];
        al[i] = aL[k0 + i];  al[i + 8] = aL[k0 + 16 + i];
      }
#pragma unroll
      for (int i = 0; i < 8; ++i) { bh[i] = bH[k0 + i]; bh[i + 8] = bH[k0 + 16 + i]; bl[i] = bL[k0 + i]; bl[i + 8] = bL[k0 + 16 + i]; }
      accv = __builtin_amdgcn_wmma_f32_16x16x32_bf16(false, ah, false, bh, (short)0, accv, false, false);
      accv = __builtin_amdgcn_wmma_f32_16x16x32_bf16(false, ah, false, bl, (short)0, accv, false, false);
      accv = __builtin_amdgcn_wmma_f32_16x16x32_bf16(false, al, false, bh, (short)0, accv, false, false);
    }
#pragma unroll
    for (int r = 0; r < 8; ++r) {
      const int i = ti * 16 + r + (uh ? 8 : 0);
      const int j = tj * 16 + mloc;
      if (i < PP && j < PP && i != j) local += expf(accv[r]);
    }
  }
#pragma unroll
  for (int q = 1; q < 32; q <<= 1) local += __shfl_xor(local, q, 32);
  if (lane == 0) red[w] = local;
  __syncthreads();
  if (tid == 0) {
    float S = 0.f;
#pragma unroll
    for (int i = 0; i < 8; ++i) S += red[i];
    atomicAdd(&acc[4], S);
  }
}

// ---------------- kernel 5: combine ----------------
__global__ void final_kernel(const float* __restrict__ acc, float* __restrict__ out) {
  const float sim  = acc[0] / (float)BB;
  const float sep  = (acc[2] > 0.5f) ? (acc[1] / acc[2]) : 0.f;
  const float ce   = acc[3] / (float)BB;
  const float text = logf(acc[4]) / (float)PP;
  out[0] = sim + ce + text + sep;   // ALPHA = BETA = 1
}

extern "C" void kernel_launch(void* const* d_in, const int* in_sizes, int n_in,
                              void* d_out, int out_size, void* d_ws, size_t ws_size,
                              hipStream_t stream) {
  const float* feat   = (const float*)d_in[0];
  const float* logits = (const float*)d_in[1];
  const int*   labels = (const int*)d_in[2];
  const float* tp     = (const float*)d_in[3];
  float* out = (float*)d_out;

  char* ws = (char*)d_ws;
  float* acc  = (float*)ws;                                          // 8 floats
  __bf16* pH  = (__bf16*)(ws + 256);                                 // [1008][512] bf16
  __bf16* pL  = (__bf16*)(ws + 256 + (size_t)PPAD * DD * 2);         // [1008][512] bf16

  init_kernel <<<1, 32, 0, stream>>>(acc);
  norm_kernel <<<PPAD, 256, 0, stream>>>(tp, pH, pL);
  ce_kernel   <<<BB, 256, 0, stream>>>(logits, labels, acc);
  sim_kernel  <<<BB / 16, 256, 0, stream>>>(feat, labels, pH, pL, acc);
  text_kernel <<<(NT * NT + 7) / 8, 256, 0, stream>>>(pH, pL, acc);
  final_kernel<<<1, 1, 0, stream>>>(acc, out);
}
